// CrossAttention_23648089932409
// MI455X (gfx1250) — matrix operand order repaired and checked by reading, not compiled, not run on hardware
//
#include <hip/hip_runtime.h>

typedef unsigned short u16;
typedef __attribute__((ext_vector_type(16))) __bf16 v16bf;
typedef __attribute__((ext_vector_type(8)))  __bf16 v8bf;
typedef __attribute__((ext_vector_type(8)))  float  v8f;
typedef __attribute__((ext_vector_type(4)))  unsigned int v4u;
typedef __attribute__((ext_vector_type(8)))  int v8i;
typedef __attribute__((ext_vector_type(4)))  int v4i;

union Acc  { v8f v; float f[8]; };
union Frag { v16bf v; struct { v8bf lo, hi; } h; };

// ---------------------------------------------------------------- helpers

__device__ __forceinline__ u16 f2bf(float x) {
    unsigned int u = __builtin_bit_cast(unsigned int, x);
    unsigned int r = u + 0x7FFFu + ((u >> 16) & 1u);   // round-to-nearest-even
    return (u16)(r >> 16);
}

// A fragment, 16x32 bf16 tile at (rowbase, kbase); stride in elements.
// lanes 0-15: row=lane, K 0..7 & 16..23 ; lanes 16-31: row=lane-16, K 8..15 & 24..31
__device__ __forceinline__ v16bf load_a_frag(const u16* base, int stride, int lane) {
    int hl = lane >> 4, r = lane & 15;
    const u16* p = base + r * stride + hl * 8;
    Frag f;
    f.h.lo = *(const v8bf*)p;
    f.h.hi = *(const v8bf*)(p + 16);
    return f.v;
}

// B fragment, 32x16 bf16 tile; row n of `base` holds column n's K-run.
// lanes 0-15: col=lane, K 0..7 & 16..23 ; lanes 16-31: K 8..15 & 24..31
__device__ __forceinline__ v16bf load_b_frag(const u16* base, int stride, int lane) {
    int hl = lane >> 4, r = lane & 15;
    const u16* p = base + r * stride + hl * 8;
    Frag f;
    f.h.lo = *(const v8bf*)p;
    f.h.hi = *(const v8bf*)(p + 16);
    return f.v;
}

__device__ __forceinline__ v8f wmma_bf16(v16bf a, v16bf b, v8f c) {
    return __builtin_amdgcn_wmma_f32_16x16x32_bf16(
        false, a, false, b, (short)0, c, false, false);
}

// ------------------------------------------------------- TDM tile load
// DMA a (tile_rows x 64) bf16 tile (row stride 1024 elements) from a
// [tensor_rows, 1024] bf16 tensor into LDS laid out with 72-u16 row stride
// (pad 4 DWORDs after every 32 DWORDs). Issue from ONE wave; pair with
// s_wait_tensorcnt before the consuming barrier.
__device__ __forceinline__ void tdm_load_tile(const u16* gsrc, void* lds_dst,
                                              unsigned tile_rows, unsigned tensor_rows) {
    unsigned long long ga = (unsigned long long)(uintptr_t)gsrc;
    unsigned ldsoff = (unsigned)(uintptr_t)lds_dst;   // LDS aperture: addr[31:0]
    v4u g0;
    g0.x = 1u;                                         // count=1 (valid descriptor)
    g0.y = ldsoff;                                     // lds_addr
    g0.z = (unsigned)(ga & 0xFFFFFFFFu);               // global_addr[31:0]
    g0.w = (unsigned)((ga >> 32) & 0x1FFFFFFu)         // global_addr[56:32]
         | (2u << 30);                                 // type = 2 (image)
    v8i g1;
    g1[0] = (int)((1u << 16)          // data_size = 1 -> 2 bytes
                | (1u << 20)          // pad_enable
                | (4u << 22)          // pad_interval code 4 -> every 32 DWORDs (128B)
                | (3u << 25));        // pad_amount  code 3 -> 4 DWORDs => 144B stride
    g1[1] = (int)(1024u << 16);       // tensor_dim0 = 1024 (low 16)
    g1[2] = (int)(tensor_rows << 16); // tensor_dim0 hi=0 | tensor_dim1 low16
    g1[3] = (int)(64u << 16);         // tensor_dim1 hi=0 | tile_dim0 = 64
    g1[4] = (int)tile_rows;           // tile_dim1 | tile_dim2 = 0
    g1[5] = 1024;                     // tensor_dim0_stride low32 (elements)
    g1[6] = 0;
    g1[7] = 0;
    v4i z4 = {0, 0, 0, 0};
#if defined(__clang_major__) && (__clang_major__ >= 23)
    v8i z8 = {0, 0, 0, 0, 0, 0, 0, 0};
    __builtin_amdgcn_tensor_load_to_lds(g0, g1, z4, z4, z8, 0);
#else
    __builtin_amdgcn_tensor_load_to_lds(g0, g1, z4, z4, 0);
#endif
}

// ------------------------------------------------------- f32 -> bf16 pass
__global__ __launch_bounds__(256) void cvt_f32_bf16(
    const float* __restrict__ src, u16* __restrict__ dst, int n) {
    int i = (blockIdx.x * 256 + threadIdx.x) * 8;
    if (i < n) {
        float4 a = *(const float4*)(src + i);
        float4 b = *(const float4*)(src + i + 4);
        u16 v[8] = { f2bf(a.x), f2bf(a.y), f2bf(a.z), f2bf(a.w),
                     f2bf(b.x), f2bf(b.y), f2bf(b.z), f2bf(b.w) };
        *(uint4*)(dst + i) = *(uint4*)v;
    }
}

// ---------------------------------------------------------------- GEMM
// C[4096,1024] = X @ W^T + bias ; X,W already bf16; W row-major => B column
// n == W row n. Block tile 128x64, 4 waves; wave w computes rows
// [w*32, w*32+32) x 64 cols. All tile movement via the Tensor Data Mover
// with double-buffered LDS: the next A/B pair DMAs while WMMAs consume the
// current pair (TDM ops from one wave complete in order; tensorcnt<=2 after
// issuing the next pair guarantees the current pair has landed).
template<bool OUT_BF16>
__global__ __launch_bounds__(128) void gemm_bias_wmma(
    const u16*  __restrict__ X,     // [4096,1024] bf16
    const u16*  __restrict__ Wb,    // [1024,1024] bf16
    const float* __restrict__ bias,
    void* __restrict__ outp)
{
    __shared__ u16 As[2][128][72];
    __shared__ u16 Bs[2][64][72];

    const int t = threadIdx.x, lane = t & 31, wv = t >> 5;
    const int m0 = blockIdx.x * 128;
    const int n0 = blockIdx.y * 64;

    Acc acc[2][4];
#pragma unroll
    for (int s = 0; s < 2; ++s)
#pragma unroll
        for (int j = 0; j < 4; ++j)
#pragma unroll
            for (int g = 0; g < 8; ++g) acc[s][j].f[g] = 0.f;

    const u16* xsrc = X  + (size_t)m0 * 1024;
    const u16* wsrc = Wb + (size_t)n0 * 1024;

    if (wv == 0) {
        tdm_load_tile(xsrc, &As[0][0][0], 128, 4096);
        tdm_load_tile(wsrc, &Bs[0][0][0], 64, 1024);
    }

    for (int i = 0; i < 16; ++i) {
        const int cur = i & 1, nxt = cur ^ 1;
        if (wv == 0) {
            if (i + 1 < 16) {
                const int k0 = (i + 1) * 64;
                tdm_load_tile(xsrc + k0, &As[nxt][0][0], 128, 4096);
                tdm_load_tile(wsrc + k0, &Bs[nxt][0][0], 64, 1024);
                __builtin_amdgcn_s_wait_tensorcnt(2);   // current pair landed
            } else {
                __builtin_amdgcn_s_wait_tensorcnt(0);
            }
        }
        __syncthreads();
#pragma unroll
        for (int kk = 0; kk < 64; kk += 32) {
            v16bf af0 = load_a_frag(&As[cur][wv * 32][kk],      72, lane);
            v16bf af1 = load_a_frag(&As[cur][wv * 32 + 16][kk], 72, lane);
#pragma unroll
            for (int j = 0; j < 4; ++j) {
                v16bf bfr = load_b_frag(&Bs[cur][j * 16][kk], 72, lane);
                acc[0][j].v = wmma_bf16(af0, bfr, acc[0][j].v);
                acc[1][j].v = wmma_bf16(af1, bfr, acc[1][j].v);
            }
        }
        __syncthreads();   // all reads of buf[cur] done before its next DMA
    }

    const int hl = lane >> 4, r = lane & 15;
#pragma unroll
    for (int s = 0; s < 2; ++s) {
        const int rowbase = m0 + wv * 32 + s * 16 + hl * 8;
#pragma unroll
        for (int j = 0; j < 4; ++j) {
            const int col = n0 + j * 16 + r;
            const float bv = bias[col];
#pragma unroll
            for (int g = 0; g < 8; ++g) {
                float res = acc[s][j].f[g] + bv;
                if (OUT_BF16)
                    ((u16*)outp)[(size_t)(rowbase + g) * 1024 + col] = f2bf(res);
                else
                    ((float*)outp)[(size_t)(rowbase + g) * 1024 + col] = res;
            }
        }
    }
}

// ---------------------------------------------------------------- attention
// Flash-style: block = (b,h) x 64 query rows; 4 waves, 16 rows each.
// Q,K,V are bf16 [B,S,1024]; head slice at col h*64. Q and K tiles are
// DMA'd into LDS by the Tensor Data Mover; V is staged transposed manually.
__global__ __launch_bounds__(128) void attn_wmma(
    const u16* __restrict__ Q, const u16* __restrict__ K,
    const u16* __restrict__ V, u16* __restrict__ O,
    const float* __restrict__ scale_p)
{
    __shared__ u16 Qs[64][72];
    __shared__ u16 Ks[64][72];
    __shared__ u16 Vs[64][72];        // transposed: Vs[d][m]
    __shared__ u16 Ps[4][16][72];     // per-wave P strip

    const int t = threadIdx.x, lane = t & 31, wv = t >> 5;
    const int hl = lane >> 4, r = lane & 15;
    const int bh = blockIdx.y;
    const int bb = bh >> 4, hh = bh & 15;
    const int n0 = blockIdx.x * 64;
    const float scale = scale_p[0];
    const u16* qbase = Q + ((size_t)bb * 2048) * 1024 + hh * 64;
    const u16* kbase = K + ((size_t)bb * 2048) * 1024 + hh * 64;
    const u16* vbase = V + ((size_t)bb * 2048) * 1024 + hh * 64;

    const int srow = t >> 1, scol = (t & 1) * 32;

    // stage Q tile once via TDM
    if (wv == 0) {
        tdm_load_tile(qbase + (size_t)n0 * 1024, &Qs[0][0], 64, 2048);
        __builtin_amdgcn_s_wait_tensorcnt(0);
    }
    __syncthreads();
    v16bf qf0 = load_a_frag(&Qs[wv * 16][0], 72, lane);
    v16bf qf1 = load_a_frag(&Qs[wv * 16][32], 72, lane);
    __syncthreads();

    Acc oacc[4];
    float rmax[8], rsum[8];
#pragma unroll
    for (int j = 0; j < 4; ++j)
#pragma unroll
        for (int g = 0; g < 8; ++g) oacc[j].f[g] = 0.f;
#pragma unroll
    for (int g = 0; g < 8; ++g) { rmax[g] = -1e30f; rsum[g] = 0.f; }

    for (int m0 = 0; m0 < 2048; m0 += 64) {
        // K tile via TDM (wave 0), V tile transposed by all threads
        if (wv == 0) tdm_load_tile(kbase + (size_t)m0 * 1024, &Ks[0][0], 64, 2048);
        {
            const u16* vsrc = vbase + (size_t)(m0 + srow) * 1024 + scol;
            u16 tmp[32];
#pragma unroll
            for (int c = 0; c < 32; c += 8) *(uint4*)&tmp[c] = *(const uint4*)(vsrc + c);
#pragma unroll
            for (int c = 0; c < 32; ++c) Vs[scol + c][srow] = tmp[c];
        }
        if (wv == 0) __builtin_amdgcn_s_wait_tensorcnt(0);
        __syncthreads();

        // S = Q @ K^T (B col m == K row m, contiguous)
        Acc sacc[4];
#pragma unroll
        for (int j = 0; j < 4; ++j)
#pragma unroll
            for (int g = 0; g < 8; ++g) sacc[j].f[g] = 0.f;
#pragma unroll
        for (int kk = 0; kk < 64; kk += 32) {
            v16bf af = (kk == 0) ? qf0 : qf1;
#pragma unroll
            for (int j = 0; j < 4; ++j) {
                v16bf bfr = load_b_frag(&Ks[j * 16][kk], 72, lane);
                sacc[j].v = wmma_bf16(af, bfr, sacc[j].v);
            }
        }

        // scale + row max (half-wave reduction: lanes 0-15 / 16-31 groups)
        float tmax[8];
#pragma unroll
        for (int g = 0; g < 8; ++g) {
            float m = sacc[0].f[g] * scale;
#pragma unroll
            for (int j = 1; j < 4; ++j) m = fmaxf(m, sacc[j].f[g] * scale);
            tmax[g] = m;
        }
#pragma unroll
        for (int off = 1; off < 16; off <<= 1)
#pragma unroll
            for (int g = 0; g < 8; ++g)
                tmax[g] = fmaxf(tmax[g], __shfl_xor(tmax[g], off, 32));

        float alpha[8];
#pragma unroll
        for (int g = 0; g < 8; ++g) {
            float mn = fmaxf(rmax[g], tmax[g]);
            alpha[g] = __expf(rmax[g] - mn);
            rmax[g] = mn;
        }

        // P = exp(scale*S - m); write per-wave strip to LDS; partial row sums
        float psum[8];
#pragma unroll
        for (int g = 0; g < 8; ++g) psum[g] = 0.f;
#pragma unroll
        for (int j = 0; j < 4; ++j)
#pragma unroll
            for (int g = 0; g < 8; ++g) {
                float p = __expf(sacc[j].f[g] * scale - rmax[g]);
                psum[g] += p;
                Ps[wv][hl * 8 + g][j * 16 + r] = f2bf(p);
            }
#pragma unroll
        for (int off = 1; off < 16; off <<= 1)
#pragma unroll
            for (int g = 0; g < 8; ++g)
                psum[g] += __shfl_xor(psum[g], off, 32);
#pragma unroll
        for (int g = 0; g < 8; ++g) rsum[g] = rsum[g] * alpha[g] + psum[g];

        // rescale O accumulators
#pragma unroll
        for (int j = 0; j < 4; ++j)
#pragma unroll
            for (int g = 0; g < 8; ++g) oacc[j].f[g] *= alpha[g];

        // O += P @ V  (B col d == Vs row d, contiguous in m)
        v16bf pf0 = load_a_frag(&Ps[wv][0][0], 72, lane);
        v16bf pf1 = load_a_frag(&Ps[wv][0][32], 72, lane);
#pragma unroll
        for (int kk = 0; kk < 64; kk += 32) {
            v16bf af = (kk == 0) ? pf0 : pf1;
#pragma unroll
            for (int jd = 0; jd < 4; ++jd) {
                v16bf bfr = load_b_frag(&Vs[jd * 16][kk], 72, lane);
                oacc[jd].v = wmma_bf16(af, bfr, oacc[jd].v);
            }
        }
        __syncthreads();
    }

    // normalize and store bf16 back into [B,S,1024] layout
    float rinv[8];
#pragma unroll
    for (int g = 0; g < 8; ++g) rinv[g] = 1.f / rsum[g];
#pragma unroll
    for (int jd = 0; jd < 4; ++jd) {
        const int col = hh * 64 + jd * 16 + r;
#pragma unroll
        for (int g = 0; g < 8; ++g) {
            const int row = n0 + wv * 16 + hl * 8 + g;
            O[((size_t)bb * 2048 + row) * 1024 + col] = f2bf(oacc[jd].f[g] * rinv[g]);
        }
    }
}

// ---------------------------------------------------------------- launch

extern "C" void kernel_launch(void* const* d_in, const int* in_sizes, int n_in,
                              void* d_out, int out_size, void* d_ws, size_t ws_size,
                              hipStream_t stream) {
    const float* query = (const float*)d_in[0];
    const float* key   = (const float*)d_in[1];
    const float* value = (const float*)d_in[2];
    const float* Wq = (const float*)d_in[3]; const float* bq = (const float*)d_in[4];
    const float* Wk = (const float*)d_in[5]; const float* bk = (const float*)d_in[6];
    const float* Wv = (const float*)d_in[7]; const float* bv = (const float*)d_in[8];
    const float* Wp = (const float*)d_in[9]; const float* bp = (const float*)d_in[10];
    const float* scale = (const float*)d_in[11];

    const size_t NX = (size_t)4096 * 1024;   // activation elements
    const size_t NW = (size_t)1024 * 1024;   // weight elements
    u16* xq = (u16*)d_ws;
    u16* xk = xq + NX;
    u16* xv = xk + NX;
    u16* wq = xv + NX;
    u16* wk = wq + NW;
    u16* wv = wk + NW;
    u16* wp = wv + NW;
    u16* qp = wp + NW;
    u16* kp = qp + NX;
    u16* vp = kp + NX;
    u16* xa = vp + NX;

    // one-pass fp32 -> bf16 (16M elements, pure bandwidth)
    const int gx = (int)(NX / 8 / 256), gw = (int)(NW / 8 / 256);
    cvt_f32_bf16<<<gx, 256, 0, stream>>>(query, xq, (int)NX);
    cvt_f32_bf16<<<gx, 256, 0, stream>>>(key,   xk, (int)NX);
    cvt_f32_bf16<<<gx, 256, 0, stream>>>(value, xv, (int)NX);
    cvt_f32_bf16<<<gw, 256, 0, stream>>>(Wq, wq, (int)NW);
    cvt_f32_bf16<<<gw, 256, 0, stream>>>(Wk, wk, (int)NW);
    cvt_f32_bf16<<<gw, 256, 0, stream>>>(Wv, wv, (int)NW);
    cvt_f32_bf16<<<gw, 256, 0, stream>>>(Wp, wp, (int)NW);

    dim3 gg(32, 16), gb(128);
    gemm_bias_wmma<true><<<gg, gb, 0, stream>>>(xq, wq, bq, qp);
    gemm_bias_wmma<true><<<gg, gb, 0, stream>>>(xk, wk, bk, kp);
    gemm_bias_wmma<true><<<gg, gb, 0, stream>>>(xv, wv, bv, vp);

    attn_wmma<<<dim3(32, 32), 128, 0, stream>>>(qp, kp, vp, xa, scale);

    gemm_bias_wmma<false><<<gg, gb, 0, stream>>>(xa, wp, bp, d_out);
}
